// LinearAttentionBlock_60112362275430
// MI455X (gfx1250) — matrix operand order repaired and checked by reading, not compiled, not run on hardware
//
#include <hip/hip_runtime.h>

typedef __attribute__((ext_vector_type(16))) __bf16        v16bf;
typedef __attribute__((ext_vector_type(8)))  float         v8f;
typedef __attribute__((ext_vector_type(4)))  unsigned int  u32x4;
typedef __attribute__((ext_vector_type(4)))  float         f32x4;

#define SCALE_Q 0.17677669529663687f  /* 32^-0.5 */

__device__ __forceinline__ unsigned short f2bf(float f) {
    unsigned u = __float_as_uint(f);
    unsigned r = u + 0x7FFFu + ((u >> 16) & 1u);   // RNE
    return (unsigned short)(r >> 16);
}
__device__ __forceinline__ float bf2f(unsigned short h) {
    return __uint_as_float(((unsigned)h) << 16);
}

union FragU { v16bf v; u32x4 q[2]; };

// A-fragment (16x32 bf16): lane<16 -> row=lane, K chunks {0..7},{16..23};
// lane>=16 -> row=lane-16, K chunks {8..15},{24..31}.  (ISA 7.12.2)
template <typename T>
__device__ __forceinline__ v16bf load_frag_a(const T* base, int stride_halfs) {
    int l = threadIdx.x & 31;
    const T* p = base + (l & 15) * stride_halfs + ((l >> 4) << 3);
    FragU f;
    f.q[0] = *(const u32x4*)(p);
    f.q[1] = *(const u32x4*)(p + 16);
    return f.v;
}
// B-fragment (32x16 bf16, loaded as N rows x K): lane<16 -> n=lane, K chunks {0..7},{16..23};
// lane>=16 -> n=lane-16, K chunks {8..15},{24..31}.
template <typename T>
__device__ __forceinline__ v16bf load_frag_b(const T* base, int stride_halfs) {
    int l = threadIdx.x & 31;
    const T* p = base + (l & 15) * stride_halfs + ((l >> 4) << 3);
    FragU f;
    f.q[0] = *(const u32x4*)(p);
    f.q[1] = *(const u32x4*)(p + 16);
    return f.v;
}

__device__ __forceinline__ v8f wmma_bf16(v16bf a, v16bf b, v8f c) {
    return __builtin_amdgcn_wmma_f32_16x16x32_bf16(false, a, false, b,
                                                   (short)0, c, false, false);
}

// ---------------------------------------------------------------- weights -> bf16
__global__ void __launch_bounds__(256)
cvt_w_kernel(const float* __restrict__ qkvw, const float* __restrict__ outw,
             unsigned short* __restrict__ wq, unsigned short* __restrict__ wo) {
    int i = blockIdx.x * 256 + threadIdx.x;
    if (i < 98304) wq[i] = f2bf(qkvw[i]);   // [384,256]
    if (i < 32768) wo[i] = f2bf(outw[i]);   // [256,128]
}

// ---------------------------------------------------------------- groupnorm stats
__global__ void __launch_bounds__(256)
gn_stats_kernel(const float* __restrict__ x, float* __restrict__ mu, float* __restrict__ rs) {
    __shared__ float s1[256], s2[256];
    int bg = blockIdx.x, tid = threadIdx.x;
    const float* p = x + (size_t)bg * 32768;
    float a = 0.f, b = 0.f;
    for (int i = tid; i < 32768; i += 256) { float v = p[i]; a += v; b += v * v; }
    s1[tid] = a; s2[tid] = b; __syncthreads();
    for (int st = 128; st > 0; st >>= 1) {
        if (tid < st) { s1[tid] += s1[tid + st]; s2[tid] += s2[tid + st]; }
        __syncthreads();
    }
    if (tid == 0) {
        float m = s1[0] * (1.f / 32768.f);
        float var = s2[0] * (1.f / 32768.f) - m * m;
        mu[bg] = m; rs[bg] = rsqrtf(var + 1e-5f);
    }
}

// ---------------------------------------------------------------- GEMM1: gn1 + qkv + epilogues
// grid (32 token-tiles of 128, 32 batches), 256 threads = 8 waves.
// Wave tiling: 2 M-halves (64 tokens = 4 M-tiles) x 4 N-groups (3 output pairs each).
// Each B fragment pair feeds 8 WMMAs (4 M-tiles x 2 N-tiles); next-k B is prefetched.
__global__ void __launch_bounds__(256)
gemm1_kernel(const float* __restrict__ x,
             const float* __restrict__ gn1w, const float* __restrict__ gn1b,
             const float* __restrict__ mu1, const float* __restrict__ rs1,
             const unsigned short* __restrict__ wq,
             unsigned short* __restrict__ qf,      // [B, n, 128] bf16 (q softmaxed * SCALE)
             float* __restrict__ kws,              // [B, 128, n] f32 raw k
             float* __restrict__ vsum) {           // [B, 4, n]  f32  sum_d v
    __shared__ unsigned short As[128 * 264];       // 67.6 KB (CDNA5: up to 320KB/WG)
    int b = blockIdx.y;
    int t0 = blockIdx.x * 128;
    int tid = threadIdx.x;
    const float* xb = x + ((size_t)b << 20);

    // phase 1: normalize 128 tokens x 256 channels -> bf16 LDS (token-major)
    for (int i = 0; i < 128; ++i) {
        int id = i * 256 + tid;
        int c = id >> 7, tok = id & 127;
        int g = b * 32 + (c >> 3);
        float v = xb[(size_t)c * 4096 + t0 + tok];
        v = (v - mu1[g]) * rs1[g] * gn1w[c] + gn1b[c];
        As[tok * 264 + c] = f2bf(v);
    }
    __syncthreads();

    int wave = tid >> 5, lane = tid & 31;
    int mh = wave & 1;            // M half: tokens mh*64 .. mh*64+63 (4 M-tiles)
    int ng = wave >> 1;           // N group: pairs 3*ng .. 3*ng+2
    int l15 = lane & 15, lh = lane >> 4;

    for (int pi = 0; pi < 3; ++pi) {
        int p = ng * 3 + pi;      // pair of N-tiles: outputs o = 32p .. 32p+31
        int nt0 = 2 * p, nt1 = nt0 + 1;
        const unsigned short* bp0 = wq + (size_t)nt0 * 16 * 256;
        const unsigned short* bp1 = wq + (size_t)nt1 * 16 * 256;
        __builtin_prefetch(bp0, 0, 1);
        v8f acc[4][2];
#pragma unroll
        for (int mi = 0; mi < 4; ++mi) {
            acc[mi][0] = (v8f){0.f,0.f,0.f,0.f,0.f,0.f,0.f,0.f};
            acc[mi][1] = acc[mi][0];
        }
        v16bf b0 = load_frag_b(bp0, 256);
        v16bf b1 = load_frag_b(bp1, 256);
#pragma unroll
        for (int k8 = 0; k8 < 8; ++k8) {
            v16bf a[4];
#pragma unroll
            for (int mi = 0; mi < 4; ++mi)
                a[mi] = load_frag_a(As + (mh * 64 + mi * 16) * 264 + k8 * 32, 264);
            v16bf nb0 = b0, nb1 = b1;
            if (k8 < 7) {                        // prefetch next-k B fragments
                nb0 = load_frag_b(bp0 + (k8 + 1) * 32, 256);
                nb1 = load_frag_b(bp1 + (k8 + 1) * 32, 256);
            }
#pragma unroll
            for (int mi = 0; mi < 4; ++mi) acc[mi][0] = wmma_bf16(a[mi], b0, acc[mi][0]);
#pragma unroll
            for (int mi = 0; mi < 4; ++mi) acc[mi][1] = wmma_bf16(a[mi], b1, acc[mi][1]);
            b0 = nb0; b1 = nb1;
        }
        if (p < 4) {                    // ---- q: softmax over d (32) per token, * SCALE
            int h = p;
#pragma unroll
            for (int mi = 0; mi < 4; ++mi) {
                int tokbase = t0 + mh * 64 + mi * 16 + lh * 8;
#pragma unroll
                for (int j = 0; j < 8; ++j) {
                    float a0 = acc[mi][0][j], a1 = acc[mi][1][j];
                    float mx = fmaxf(a0, a1);
                    mx = fmaxf(mx, __shfl_xor(mx, 1));
                    mx = fmaxf(mx, __shfl_xor(mx, 2));
                    mx = fmaxf(mx, __shfl_xor(mx, 4));
                    mx = fmaxf(mx, __shfl_xor(mx, 8));
                    float e0 = __expf(a0 - mx), e1 = __expf(a1 - mx);
                    float sm = e0 + e1;
                    sm += __shfl_xor(sm, 1);
                    sm += __shfl_xor(sm, 2);
                    sm += __shfl_xor(sm, 4);
                    sm += __shfl_xor(sm, 8);
                    float inv = SCALE_Q / sm;
                    size_t ro = ((size_t)b * 4096 + tokbase + j) * 128 + h * 32 + l15;
                    qf[ro]      = f2bf(e0 * inv);
                    qf[ro + 16] = f2bf(e1 * inv);
                }
            }
        } else if (p < 8) {             // ---- k: store raw f32, d-major
            int hd0 = (p - 4) * 32 + l15;
#pragma unroll
            for (int mi = 0; mi < 4; ++mi) {
                int tokbase = t0 + mh * 64 + mi * 16 + lh * 8;
#pragma unroll
                for (int j = 0; j < 8; ++j) {
                    int tok = tokbase + j;
                    kws[((size_t)b * 128 + hd0)      * 4096 + tok] = acc[mi][0][j];
                    kws[((size_t)b * 128 + hd0 + 16) * 4096 + tok] = acc[mi][1][j];
                }
            }
        } else {                        // ---- v: vsum over 32 dims per token
            int h = p - 8;
#pragma unroll
            for (int mi = 0; mi < 4; ++mi) {
                int tokbase = t0 + mh * 64 + mi * 16 + lh * 8;
#pragma unroll
                for (int j = 0; j < 8; ++j) {
                    float v = acc[mi][0][j] + acc[mi][1][j];
                    v += __shfl_xor(v, 1);
                    v += __shfl_xor(v, 2);
                    v += __shfl_xor(v, 4);
                    v += __shfl_xor(v, 8);
                    if (l15 == 0)
                        vsum[((size_t)b * 4 + h) * 4096 + tokbase + j] = v;
                }
            }
        }
    }
}

// ---------------------------------------------------------------- k softmax (over tokens) dot vsum
// one block per (b, hd): s[b,hd] = sum_n softmax_n(k)[n] * vsum[n]  (incl. 4 mem tokens)
__global__ void __launch_bounds__(256)
ksm_kernel(const float* __restrict__ kws, const float* __restrict__ vsum,
           const float* __restrict__ memkv, float* __restrict__ sout) {
    __shared__ float r1[256], r2[256];
    int bh = blockIdx.x, tid = threadIdx.x;
    int b = bh >> 7, hd = bh & 127, h = hd >> 5, d = hd & 31;
    const float* kp = kws + (size_t)bh * 4096;
    const float* vp = vsum + ((size_t)b * 4 + h) * 4096;

    float mx = -3.4e38f;
    for (int i = tid; i < 4096; i += 256) mx = fmaxf(mx, kp[i]);
    if (tid < 4) mx = fmaxf(mx, memkv[(h * 32 + d) * 4 + tid]);
    r1[tid] = mx; __syncthreads();
    for (int st = 128; st > 0; st >>= 1) {
        if (tid < st) r1[tid] = fmaxf(r1[tid], r1[tid + st]);
        __syncthreads();
    }
    mx = r1[0]; __syncthreads();

    float es = 0.f, ws = 0.f;
    for (int i = tid; i < 4096; i += 256) {
        float e = __expf(kp[i] - mx);
        es += e; ws += e * vp[i];
    }
    if (tid < 4) {   // memory tokens: mk = mem_kv[0,h,d,j], mv = sum_e mem_kv[1,h,e,j]
        float mk = memkv[(h * 32 + d) * 4 + tid];
        float mv = 0.f;
        for (int e = 0; e < 32; ++e) mv += memkv[512 + h * 128 + e * 4 + tid];
        float ee = __expf(mk - mx);
        es += ee; ws += ee * mv;
    }
    r1[tid] = es; r2[tid] = ws; __syncthreads();
    for (int st = 128; st > 0; st >>= 1) {
        if (tid < st) { r1[tid] += r1[tid + st]; r2[tid] += r2[tid + st]; }
        __syncthreads();
    }
    if (tid == 0) sout[bh] = r2[0] / r1[0];
}

// ---------------------------------------------------------------- GEMM2: (s*q) @ out_w^T + out_b
// Wave tiling: 2 M-halves (64 tokens = 4 M-tiles) x 4 N-groups (4 channel tiles each).
__global__ void __launch_bounds__(256)
gemm2_kernel(const unsigned short* __restrict__ qf, const float* __restrict__ s,
             const unsigned short* __restrict__ wo, const float* __restrict__ outb,
             float* __restrict__ out) {
    __shared__ unsigned short As[128 * 136];   // 34.8 KB
    __shared__ float ot[8][16][17];
    int b = blockIdx.y, t0 = blockIdx.x * 128, tid = threadIdx.x;

    // phase 1: A[token, hd] = qf * s[b,hd]  (bf16)
    for (int i = 0; i < 64; ++i) {
        int id = i * 256 + tid;
        int tok = id >> 7, hd = id & 127;
        float v = bf2f(qf[((size_t)b * 4096 + t0 + tok) * 128 + hd]) * s[b * 128 + hd];
        As[tok * 136 + hd] = f2bf(v);
    }
    __syncthreads();

    int wave = tid >> 5, lane = tid & 31;
    int mh = wave & 1, ng = wave >> 1;
    int l15 = lane & 15, lh = lane >> 4;

    for (int ni = 0; ni < 4; ++ni) {
        int nt = ng * 4 + ni;              // output-channel tile (16 of 256)
        const unsigned short* bp = wo + (size_t)nt * 16 * 128;
        v8f acc[4];
#pragma unroll
        for (int mi = 0; mi < 4; ++mi)
            acc[mi] = (v8f){0.f,0.f,0.f,0.f,0.f,0.f,0.f,0.f};
        v16bf bw = load_frag_b(bp, 128);
#pragma unroll
        for (int k8 = 0; k8 < 4; ++k8) {
            v16bf a[4];
#pragma unroll
            for (int mi = 0; mi < 4; ++mi)
                a[mi] = load_frag_a(As + (mh * 64 + mi * 16) * 136 + k8 * 32, 136);
            v16bf nb = bw;
            if (k8 < 3) nb = load_frag_b(bp + (k8 + 1) * 32, 128);
#pragma unroll
            for (int mi = 0; mi < 4; ++mi)
                acc[mi] = wmma_bf16(a[mi], bw, acc[mi]);
            bw = nb;
        }
        float bias = outb[nt * 16 + l15];
#pragma unroll
        for (int mi = 0; mi < 4; ++mi) {
#pragma unroll
            for (int j = 0; j < 8; ++j)
                ot[wave][lh * 8 + j][l15] = acc[mi][j] + bias;
            __syncthreads();
            // coalesced transposed store: each lane writes 8 consecutive tokens
            int ocl = lane >> 1;
            int tseg = (lane & 1) * 8;
            f32x4 v0, v1;
            v0[0] = ot[wave][tseg + 0][ocl]; v0[1] = ot[wave][tseg + 1][ocl];
            v0[2] = ot[wave][tseg + 2][ocl]; v0[3] = ot[wave][tseg + 3][ocl];
            v1[0] = ot[wave][tseg + 4][ocl]; v1[1] = ot[wave][tseg + 5][ocl];
            v1[2] = ot[wave][tseg + 6][ocl]; v1[3] = ot[wave][tseg + 7][ocl];
            size_t base = ((size_t)b * 256 + nt * 16 + ocl) * 4096
                        + t0 + mh * 64 + mi * 16 + tseg;
            *(f32x4*)(out + base)     = v0;
            *(f32x4*)(out + base + 4) = v1;
            __syncthreads();
        }
    }
}

// ---------------------------------------------------------------- groupnorm2 apply (in place)
__global__ void __launch_bounds__(256)
gn_apply_kernel(float* __restrict__ o, const float* __restrict__ mu, const float* __restrict__ rs,
                const float* __restrict__ w, const float* __restrict__ bb) {
    size_t idx = (size_t)blockIdx.x * 256 + threadIdx.x;   // float4 index, 8388608 total
    int c = (int)((idx >> 10) & 255);
    int bg = (int)(idx >> 18) * 32 + (c >> 3);
    float r = rs[bg] * w[c];
    float bias = bb[c] - mu[bg] * r;
    f32x4 v = ((const f32x4*)o)[idx];
    v[0] = v[0] * r + bias; v[1] = v[1] * r + bias;
    v[2] = v[2] * r + bias; v[3] = v[3] * r + bias;
    ((f32x4*)o)[idx] = v;
}

extern "C" void kernel_launch(void* const* d_in, const int* in_sizes, int n_in,
                              void* d_out, int out_size, void* d_ws, size_t ws_size,
                              hipStream_t stream) {
    (void)in_sizes; (void)n_in; (void)out_size; (void)ws_size;
    const float* x     = (const float*)d_in[0];
    const float* gn1w  = (const float*)d_in[1];
    const float* gn1b  = (const float*)d_in[2];
    const float* qkvw  = (const float*)d_in[3];
    const float* memkv = (const float*)d_in[4];
    const float* outw  = (const float*)d_in[5];
    const float* outb  = (const float*)d_in[6];
    const float* gn2w  = (const float*)d_in[7];
    const float* gn2b  = (const float*)d_in[8];
    float* out = (float*)d_out;

    char* wsb = (char*)d_ws;
    size_t off = 0;
    auto take = [&](size_t bytes) -> void* {
        off = (off + 255) & ~(size_t)255;
        void* p = wsb + off;
        off += bytes;
        return p;
    };
    float* mu1  = (float*)take(1024 * 4);
    float* rs1  = (float*)take(1024 * 4);
    float* mu2  = (float*)take(1024 * 4);
    float* rs2  = (float*)take(1024 * 4);
    float* sctx = (float*)take(4096 * 4);
    float* vsum = (float*)take((size_t)32 * 4 * 4096 * 4);
    unsigned short* wq = (unsigned short*)take((size_t)98304 * 2);
    unsigned short* wo = (unsigned short*)take((size_t)32768 * 2);
    unsigned short* qf = (unsigned short*)take((size_t)16777216 * 2);
    float* kws = (float*)take((size_t)16777216 * 4);

    cvt_w_kernel<<<384, 256, 0, stream>>>(qkvw, outw, wq, wo);
    gn_stats_kernel<<<1024, 256, 0, stream>>>(x, mu1, rs1);
    gemm1_kernel<<<dim3(32, 32), 256, 0, stream>>>(x, gn1w, gn1b, mu1, rs1, wq, qf, kws, vsum);
    ksm_kernel<<<4096, 256, 0, stream>>>(kws, vsum, memkv, sctx);
    gemm2_kernel<<<dim3(32, 32), 256, 0, stream>>>(qf, sctx, wo, outb, out);
    gn_stats_kernel<<<1024, 256, 0, stream>>>(out, mu2, rs2);
    gn_apply_kernel<<<32768, 256, 0, stream>>>(out, mu2, rs2, gn2w, gn2b);
}
